// TtSelfAttention_8589934592339
// MI455X (gfx1250) — matrix operand order repaired and checked by reading, not compiled, not run on hardware
//
#include <hip/hip_runtime.h>

// ---------------------------------------------------------------------------
// MI455X (gfx1250) self-attention, f16 WMMA path.
//   B=2, S=2048, D=1024, H=16, dh=64.
// All matmuls use v_wmma_f32_16x16x32_f16 (f32 accumulate).
// A operands row-major [M][K]; B operands stored transposed [N][K] so every
// per-lane fragment load is a contiguous global_load_b128 per the CDNA5 ISA
// VGPR layouts (CDNA5 ISA §7.12.2).
// ---------------------------------------------------------------------------

typedef _Float16 v8h  __attribute__((ext_vector_type(8)));
typedef _Float16 v16h __attribute__((ext_vector_type(16)));
typedef float    v8f  __attribute__((ext_vector_type(8)));

union HFrag { v16h v; v8h h[2]; };

// A-matrix 16x32 f16 fragment: lanes 0-15 -> M=lane, K = {g*8+0..7, 16+g*8+0..7}
__device__ __forceinline__ v16h load_a_frag(const _Float16* base, int lda, int lane) {
  const int g = lane >> 4, m = lane & 15;
  const _Float16* p = base + (size_t)m * lda + g * 8;
  HFrag f;
  f.h[0] = *(const v8h*)(p);        // VGPR0..3 : K = g*8 + 0..7
  f.h[1] = *(const v8h*)(p + 16);   // VGPR4..7 : K = 16 + g*8 + 0..7
  return f.v;
}

// B-matrix 32x16 f16 fragment from Bt (= B^T, row-major [N][K]):
// lane = N (mod 16), K = {g*8+0..7, 16+g*8+0..7} (same order as A).
__device__ __forceinline__ v16h load_b_frag(const _Float16* base, int ldb, int lane) {
  const int g = lane >> 4, n = lane & 15;
  const _Float16* p = base + (size_t)n * ldb + g * 8;
  HFrag f;
  f.h[0] = *(const v8h*)(p);        // VGPR0..3 : K = g*8 + 0..7
  f.h[1] = *(const v8h*)(p + 16);   // VGPR4..7 : K = 16 + g*8 + 0..7
  return f.v;
}

__device__ __forceinline__ v8f wmma32(v16h a, v16h b, v8f c) {
  return __builtin_amdgcn_wmma_f32_16x16x32_f16(
      /*neg_a=*/false, a, /*neg_b=*/false, b,
      /*c_mod=*/(short)0, c, /*reuse_a=*/false, /*reuse_b=*/false);
}

// ---------------------------------------------------------------------------
// f32 -> f16 convert (grid-stride)
// ---------------------------------------------------------------------------
__global__ void cvt_f32_f16(const float* __restrict__ src,
                            _Float16* __restrict__ dst, size_t n) {
  for (size_t i = (size_t)blockIdx.x * blockDim.x + threadIdx.x; i < n;
       i += (size_t)gridDim.x * blockDim.x)
    dst[i] = (_Float16)src[i];
}

// src row-major [K][N] f32  ->  dst row-major [N][K] f16   (coalesced reads)
__global__ void transpose_cvt(const float* __restrict__ src,
                              _Float16* __restrict__ dst, int K, int N) {
  const size_t total = (size_t)K * N;
  for (size_t i = (size_t)blockIdx.x * blockDim.x + threadIdx.x; i < total;
       i += (size_t)gridDim.x * blockDim.x) {
    size_t k = i / N, n = i % N;
    dst[n * (size_t)K + k] = (_Float16)src[i];
  }
}

// ---------------------------------------------------------------------------
// Per-head V transpose: Vh [bh][s][dh] (f16) -> Vt [bh][dh][s] (f16).
// LDS-tiled 64x64 so both global read and global write are 32B-per-thread
// coalesced (avoids the 2-byte scattered stores a direct transpose would do).
// Grid = (B*H) * (S/64) blocks, 256 threads.
// ---------------------------------------------------------------------------
__global__ __launch_bounds__(256) void transpose_v(
    const _Float16* __restrict__ Vh, _Float16* __restrict__ Vt) {
  __shared__ _Float16 tile[64][72];                    // pad 72 to dodge banks
  const int t = threadIdx.x;
  const int bh = blockIdx.x >> 5;                      // 32 s-tiles per head
  const int s0 = (blockIdx.x & 31) * 64;

  // load: thread t covers row sl = t/4, cols d0..d0+15 (two v8h = 32B)
  {
    const int sl = t >> 2, d0 = (t & 3) * 16;
    const _Float16* src = Vh + ((size_t)bh * 2048 + s0 + sl) * 64 + d0;
    v8h a = *(const v8h*)(src);
    v8h b = *(const v8h*)(src + 8);
#pragma unroll
    for (int j = 0; j < 8; ++j) tile[sl][d0 + j] = a[j];
#pragma unroll
    for (int j = 0; j < 8; ++j) tile[sl][d0 + 8 + j] = b[j];
  }
  __syncthreads();
  // store: thread t covers dh row dd = t/4, s-cols sl0..sl0+15 (32B)
  {
    const int dd = t >> 2, sl0 = (t & 3) * 16;
    _Float16* dst = Vt + ((size_t)bh * 64 + dd) * 2048 + s0 + sl0;
    v8h a, b;
#pragma unroll
    for (int j = 0; j < 8; ++j) a[j] = tile[sl0 + j][dd];
#pragma unroll
    for (int j = 0; j < 8; ++j) b[j] = tile[sl0 + 8 + j][dd];
    *(v8h*)(dst) = a;
    *(v8h*)(dst + 8) = b;
  }
}

// ---------------------------------------------------------------------------
// Generic f16 WMMA GEMM:  C[M,N] = A[M,K] * Bt[N,K]^T + bias
// Block = 256 thr = 8 waves, wave tile 64x64, block tile 128x256.
// mode 0: scatter epilogue into Q [bh][s][dh], K [bh][s][dh], Vh [bh][s][dh]
//         (all three coalesced: 16 lanes x 2B contiguous = 32B stores)
// mode 1: f32 epilogue into Of32 [M][N]
// ---------------------------------------------------------------------------
__global__ __launch_bounds__(256) void gemm_wmma_f16(
    const _Float16* __restrict__ A, const _Float16* __restrict__ Bt,
    const float* __restrict__ bias, int M, int N, int K, int mode,
    _Float16* __restrict__ Qb, _Float16* __restrict__ Kb,
    _Float16* __restrict__ Vb, float* __restrict__ Of32) {
  const int lane = threadIdx.x & 31;
  const int wid  = threadIdx.x >> 5;
  const int wm   = wid & 1, wn = wid >> 1;            // 2 x 4 wave grid
  const int row0 = blockIdx.x * 128 + wm * 64;
  const int col0 = blockIdx.y * 256 + wn * 64;
  const int g = lane >> 4, ln = lane & 15;

  v8f acc[4][4];
#pragma unroll
  for (int mt = 0; mt < 4; ++mt)
#pragma unroll
    for (int nt = 0; nt < 4; ++nt) acc[mt][nt] = (v8f)(0.0f);

  for (int k0 = 0; k0 < K; k0 += 32) {
    v16h af[4], bf[4];
#pragma unroll
    for (int mt = 0; mt < 4; ++mt)
      af[mt] = load_a_frag(A + (size_t)(row0 + mt * 16) * K + k0, K, lane);
#pragma unroll
    for (int nt = 0; nt < 4; ++nt)
      bf[nt] = load_b_frag(Bt + (size_t)(col0 + nt * 16) * K + k0, K, lane);
#pragma unroll
    for (int mt = 0; mt < 4; ++mt)
#pragma unroll
      for (int nt = 0; nt < 4; ++nt)
        acc[mt][nt] = wmma32(af[mt], bf[nt], acc[mt][nt]);
  }

  float bcol[4];
#pragma unroll
  for (int nt = 0; nt < 4; ++nt) bcol[nt] = bias[col0 + nt * 16 + ln];

#pragma unroll
  for (int mt = 0; mt < 4; ++mt)
#pragma unroll
    for (int nt = 0; nt < 4; ++nt)
#pragma unroll
      for (int r = 0; r < 8; ++r) {
        const int row = row0 + mt * 16 + g * 8 + r;   // C layout: M = r + g*8
        const int col = col0 + nt * 16 + ln;          // N = lane%16
        const float v = acc[mt][nt][r] + bcol[nt];
        if (mode == 0) {
          // row = b*2048 + s ; col in [0,3072): sec (q/k/v), head, dh
          const int b = row >> 11, s = row & 2047;
          const int sec = col >> 10, d = col & 1023;
          const int h = d >> 6, dd = d & 63;
          const size_t bh = (size_t)(b * 16 + h);
          const _Float16 hv = (_Float16)v;
          const size_t idx = (bh * 2048 + s) * 64 + dd;
          if (sec == 0)      Qb[idx] = hv;
          else if (sec == 1) Kb[idx] = hv;
          else               Vb[idx] = hv;
        } else {
          Of32[(size_t)row * N + col] = v;
        }
      }
}

// ---------------------------------------------------------------------------
// Flash attention core. One wave = 32 q rows x dh=64. Block = 8 waves = 256 q.
// Grid = B*H*(S/256) = 256 blocks. KV tiles of 64, online softmax in f32.
// P (probabilities) round-trips through per-wave LDS to convert C-layout ->
// A-layout for the PV WMMA (guarded with s_wait_dscnt).
// ---------------------------------------------------------------------------
__global__ __launch_bounds__(256) void flash_attn_wmma(
    const _Float16* __restrict__ Q, const _Float16* __restrict__ Kb,
    const _Float16* __restrict__ Vt, _Float16* __restrict__ ctx) {
  __shared__ _Float16 lds[8][32 * 64];                 // 4 KB per wave

  const int lane = threadIdx.x & 31, wid = threadIdx.x >> 5;
  const int g = lane >> 4, ln = lane & 15;
  const int bh = blockIdx.x >> 3, qc = blockIdx.x & 7; // bh = b*16 + h
  const int q0 = qc * 256 + wid * 32;

  const _Float16* Qp = Q  + ((size_t)bh * 2048 + q0) * 64;
  const _Float16* Kp = Kb + (size_t)bh * 2048 * 64;
  const _Float16* Vp = Vt + (size_t)bh * 64 * 2048;

  // Q fragments: 2 m-tiles x 2 k-chunks (dh = 64 = 2 x 32)
  v16h qf[2][2];
#pragma unroll
  for (int mt = 0; mt < 2; ++mt)
#pragma unroll
    for (int kk = 0; kk < 2; ++kk)
      qf[mt][kk] = load_a_frag(Qp + (size_t)(mt * 16) * 64 + kk * 32, 64, lane);

  v8f of[2][4];
#pragma unroll
  for (int mt = 0; mt < 2; ++mt)
#pragma unroll
    for (int nt = 0; nt < 4; ++nt) of[mt][nt] = (v8f)(0.0f);
  float mi[2][8], li[2][8];
#pragma unroll
  for (int mt = 0; mt < 2; ++mt)
#pragma unroll
    for (int r = 0; r < 8; ++r) { mi[mt][r] = -1e30f; li[mt][r] = 0.0f; }

  const float scale = 0.125f;                          // 1/sqrt(64)

  for (int kt = 0; kt < 32; ++kt) {
    // --- S = Q * K^T (K stored [pos][dh] == Bt for this product) ----------
    v16h kf[2][4];
#pragma unroll
    for (int kk = 0; kk < 2; ++kk)
#pragma unroll
      for (int nt = 0; nt < 4; ++nt)
        kf[kk][nt] = load_b_frag(
            Kp + (size_t)(kt * 64 + nt * 16) * 64 + kk * 32, 64, lane);

    v8f s[2][4];
#pragma unroll
    for (int mt = 0; mt < 2; ++mt)
#pragma unroll
      for (int nt = 0; nt < 4; ++nt) {
        v8f a = (v8f)(0.0f);
#pragma unroll
        for (int kk = 0; kk < 2; ++kk) a = wmma32(qf[mt][kk], kf[kk][nt], a);
        s[mt][nt] = a * scale;
      }

    // --- online softmax ---------------------------------------------------
#pragma unroll
    for (int mt = 0; mt < 2; ++mt) {
      float nm[8], al[8];
#pragma unroll
      for (int r = 0; r < 8; ++r) {
        float mx = -1e30f;
#pragma unroll
        for (int nt = 0; nt < 4; ++nt) mx = fmaxf(mx, s[mt][nt][r]);
        // row spans the 16 lanes of this half-wave (xor masks stay in group)
        mx = fmaxf(mx, __shfl_xor(mx, 1));
        mx = fmaxf(mx, __shfl_xor(mx, 2));
        mx = fmaxf(mx, __shfl_xor(mx, 4));
        mx = fmaxf(mx, __shfl_xor(mx, 8));
        const float m_new = fmaxf(mi[mt][r], mx);
        al[r] = __expf(mi[mt][r] - m_new);
        mi[mt][r] = m_new;
        nm[r] = m_new;
      }
#pragma unroll
      for (int r = 0; r < 8; ++r) {
        float rs = 0.0f;
#pragma unroll
        for (int nt = 0; nt < 4; ++nt) {
          const float p = __expf(s[mt][nt][r] - nm[r]);
          s[mt][nt][r] = p;
          rs += p;
        }
        rs += __shfl_xor(rs, 1);
        rs += __shfl_xor(rs, 2);
        rs += __shfl_xor(rs, 4);
        rs += __shfl_xor(rs, 8);
        li[mt][r] = li[mt][r] * al[r] + rs;
      }
#pragma unroll
      for (int nt = 0; nt < 4; ++nt)
#pragma unroll
        for (int r = 0; r < 8; ++r) of[mt][nt][r] *= al[r];
      // P tile -> per-wave LDS (C layout -> row-major 32x64)
#pragma unroll
      for (int nt = 0; nt < 4; ++nt)
#pragma unroll
        for (int r = 0; r < 8; ++r)
          lds[wid][(mt * 16 + g * 8 + r) * 64 + nt * 16 + ln] =
              (_Float16)s[mt][nt][r];
    }
    asm volatile("s_wait_dscnt 0" ::: "memory");       // LDS store->load, same wave

    // --- O += P * V  (Vt stored [dh][pos] == Bt for this product) ---------
#pragma unroll
    for (int kk = 0; kk < 2; ++kk) {
      v16h pf[2], vf[4];
#pragma unroll
      for (int mt = 0; mt < 2; ++mt)
        pf[mt] = load_a_frag(&lds[wid][(mt * 16) * 64 + kk * 32], 64, lane);
#pragma unroll
      for (int nt = 0; nt < 4; ++nt)
        vf[nt] = load_b_frag(
            Vp + (size_t)(nt * 16) * 2048 + kt * 64 + kk * 32, 2048, lane);
#pragma unroll
      for (int mt = 0; mt < 2; ++mt)
#pragma unroll
        for (int nt = 0; nt < 4; ++nt)
          of[mt][nt] = wmma32(pf[mt], vf[nt], of[mt][nt]);
    }
  }

  // --- epilogue: O / l, scatter to ctx [b*2048+s][h*64+dh] (f16) ----------
  const int b = bh >> 4, h = bh & 15;
#pragma unroll
  for (int mt = 0; mt < 2; ++mt)
#pragma unroll
    for (int nt = 0; nt < 4; ++nt)
#pragma unroll
      for (int r = 0; r < 8; ++r) {
        const int qq = q0 + mt * 16 + g * 8 + r;
        const int d  = nt * 16 + ln;
        const float v = of[mt][nt][r] / li[mt][r];
        ctx[((size_t)(b * 2048 + qq)) * 1024 + h * 64 + d] = (_Float16)v;
      }
}

// ---------------------------------------------------------------------------
// Launch
// ---------------------------------------------------------------------------
extern "C" void kernel_launch(void* const* d_in, const int* in_sizes, int n_in,
                              void* d_out, int out_size, void* d_ws,
                              size_t ws_size, hipStream_t stream) {
  const float* x     = (const float*)d_in[0];   // [1,2,2048,1024]
  const float* W_in  = (const float*)d_in[1];   // [1024,3072]
  const float* b_in  = (const float*)d_in[2];   // [3072]
  const float* W_out = (const float*)d_in[3];   // [1024,1024]
  const float* b_out = (const float*)d_in[4];   // [1024]
  float* out = (float*)d_out;                   // [1,2,2048,1024] f32

  constexpr int B = 2, S = 2048, D = 1024, H = 16, dh = 64;
  constexpr size_t MS = (size_t)B * S;          // 4096 rows

  char* ws = (char*)d_ws;
  _Float16* xh    = (_Float16*)ws; ws += MS * D * 2;            // 8 MB
  _Float16* WtIn  = (_Float16*)ws; ws += (size_t)3 * D * D * 2; // 6 MB  [3D][D]
  _Float16* WtOut = (_Float16*)ws; ws += (size_t)D * D * 2;     // 2 MB  [D][D]
  _Float16* Qb    = (_Float16*)ws; ws += (size_t)B * H * S * dh * 2;
  _Float16* Kb    = (_Float16*)ws; ws += (size_t)B * H * S * dh * 2;
  _Float16* Vb    = (_Float16*)ws; ws += (size_t)B * H * S * dh * 2;
  _Float16* Vtb   = (_Float16*)ws; ws += (size_t)B * H * dh * S * 2;
  _Float16* ctxh  = (_Float16*)ws; ws += MS * D * 2;

  cvt_f32_f16<<<2048, 256, 0, stream>>>(x, xh, MS * D);
  transpose_cvt<<<4096, 256, 0, stream>>>(W_in, WtIn, D, 3 * D);
  transpose_cvt<<<2048, 256, 0, stream>>>(W_out, WtOut, D, D);

  // QKV projection: [4096,1024] x [1024,3072]
  dim3 g1(4096 / 128, 3072 / 256);
  gemm_wmma_f16<<<g1, 256, 0, stream>>>(xh, WtIn, b_in, 4096, 3 * D, D,
                                        /*mode=*/0, Qb, Kb, Vb, nullptr);

  // V -> V^T per head (LDS-tiled, fully coalesced both directions)
  transpose_v<<<B * H * (S / 64), 256, 0, stream>>>(Vb, Vtb);

  // Flash attention: 2*16 heads x 8 q-chunks of 256
  flash_attn_wmma<<<B * H * (S / 256), 256, 0, stream>>>(Qb, Kb, Vtb, ctxh);

  // Output projection: [4096,1024] x [1024,1024] -> f32 + bias
  dim3 g2(4096 / 128, 1024 / 256);
  gemm_wmma_f16<<<g2, 256, 0, stream>>>(ctxh, WtOut, b_out, 4096, D, D,
                                        /*mode=*/1, nullptr, nullptr, nullptr,
                                        out);
}
